// ModuleCorrelationTranspose_17282948399397
// MI455X (gfx1250) — matrix operand order repaired and checked by reading, not compiled, not run on hardware
//
#include <hip/hip_runtime.h>

// Correlation-transpose (adjoint of local correlation) as banded GEMM on CDNA5 WMMA.
// out[b,c,y,x] = sum_{dy,dx in [-4,4]} corr[b,(dy+4)*9+(dx+4), y-dy, x-dx] * feat[b,c,y-dy,x-dx]

#define RAD   4
#define ND    9                 // 2R+1
#define D2    81                // ND*ND corr channels
#define NT    16                // output x-tile (WMMA N)
#define CT    16                // channel tile (WMMA M)
#define KROW  24                // NT + 2*RAD source columns per dy row
#define KUSE  216               // ND * KROW
#define KP    224               // KUSE padded to multiple of 32 (7 WMMA steps)
#define CH    128               // channels staged per LDS pass (2 passes for C=256)
#define BLK   256               // 8 waves (wave32)

typedef __attribute__((ext_vector_type(16))) __bf16 v16bf;
typedef __attribute__((ext_vector_type(8)))  __bf16 v8bf;
typedef __attribute__((ext_vector_type(8)))  float  v8f;

__global__ __launch_bounds__(BLK)
void corr_transpose_wmma(const float* __restrict__ corr,   // [B, 81, H, W]
                         const float* __restrict__ feat,   // [B, C,  H, W]
                         float* __restrict__ out,          // [B, C,  H, W]
                         int B, int C, int H, int W)
{
    __shared__ __align__(16) __bf16 Ash[CH * KP];  // features, K contiguous
    __shared__ __align__(16) __bf16 Bsh[NT * KP];  // banded corr weights, K contiguous

    const int xtiles = W / NT;
    const int wg = blockIdx.x;
    const int xt = wg % xtiles;
    const int y  = (wg / xtiles) % H;
    const int b  = wg / (xtiles * H);
    const int x0 = xt * NT;

    const int tid = threadIdx.x;

    // ---------------- stage banded B matrix (shared by all c-tiles) ----------------
    // Bsh[n][dyi*KROW + k] = corr[b, dyi*9 + (n+8-k), y+4-dyi, x0-4+k] inside band, else 0.
    {
        const float* corr_b = corr + (size_t)b * D2 * H * W;
        for (int e = tid; e < NT * KP; e += BLK) {
            int n  = e / KP;
            int kk = e - n * KP;
            float v = 0.0f;
            if (kk < KUSE) {
                int dyi = kk / KROW;
                int k   = kk - dyi * KROW;
                int dxn = n + 8 - k;             // dx + 4
                int ys  = y + RAD - dyi;
                int xs  = x0 - RAD + k;
                if ((unsigned)dxn <= 8u && (unsigned)ys < (unsigned)H && (unsigned)xs < (unsigned)W)
                    v = corr_b[(size_t)(dyi * ND + dxn) * H * W + (size_t)ys * W + xs];
            }
            Bsh[n * KP + kk] = (__bf16)v;
        }
    }

    const int lane = tid & 31;          // wave32
    const int wv   = tid >> 5;          // 0..7
    const int half = lane >> 4;         // 0: lanes 0-15, 1: lanes 16-31
    const int m    = lane & 15;

    for (int h = 0; h < 2; ++h) {
        const int cbase = h * CH;
        __syncthreads();                // Bsh ready (h=0); Ash free of readers (h=1)

        // ------------- stage A matrix: 128 channels of features as bf16 -------------
        // Ash[c][dyi*KROW + k] = feat[b, cbase+c, y+4-dyi, x0-4+k], zero out-of-bounds.
        {
            const float* feat_b = feat + ((size_t)b * C + cbase) * H * W;
            for (int e = tid; e < CH * KP; e += BLK) {
                int c  = e / KP;
                int kk = e - c * KP;
                float v = 0.0f;
                if (kk < KUSE) {
                    int dyi = kk / KROW;
                    int k   = kk - dyi * KROW;
                    int ys  = y + RAD - dyi;
                    int xs  = x0 - RAD + k;
                    if ((unsigned)ys < (unsigned)H && (unsigned)xs < (unsigned)W)
                        v = feat_b[(size_t)c * H * W + (size_t)ys * W + xs];
                }
                Ash[c * KP + kk] = (__bf16)v;
            }
        }
        __syncthreads();

        // ------------- WMMA: each wave computes one 16(M=c) x 16(N=x) tile -------------
        v8f acc = {};
        const __bf16* arow = &Ash[(wv * CT + m) * KP];   // A row M=m of this wave's c-tile
        const __bf16* brow = &Bsh[m * KP];               // B column N=m
        #pragma unroll
        for (int kb = 0; kb < KP / 32; ++kb) {
            // ISA 16-bit A layout: lanes 0-15 hold K{0..7,16..23}, lanes 16-31 K{8..15,24..31}
            v8bf a0 = *(const v8bf*)(arow + kb * 32 + half * 8);
            v8bf a1 = *(const v8bf*)(arow + kb * 32 + half * 8 + 16);
            // B layout matches A: lanes 0-15 hold K{0..7,16..23} of column N, lanes 16-31 K{8..15,24..31}
            v8bf b0 = *(const v8bf*)(brow + kb * 32 + half * 8);
            v8bf b1 = *(const v8bf*)(brow + kb * 32 + half * 8 + 16);
            v16bf a  = __builtin_shufflevector(a0, a1, 0,1,2,3,4,5,6,7,8,9,10,11,12,13,14,15);
            v16bf bb = __builtin_shufflevector(b0, b1, 0,1,2,3,4,5,6,7,8,9,10,11,12,13,14,15);
            acc = __builtin_amdgcn_wmma_f32_16x16x32_bf16(
                false, a, false, bb, (short)0, acc, false, false);
        }

        // ------------- store: D VGPR r -> row M = r + 8*half, col N = m -------------
        const int cg = cbase + wv * CT + half * 8;
        float* orow = out + (((size_t)b * C + cg) * H + y) * W + x0 + m;
        #pragma unroll
        for (int r = 0; r < 8; ++r)
            orow[(size_t)r * H * W] = acc[r];
    }
}

extern "C" void kernel_launch(void* const* d_in, const int* in_sizes, int n_in,
                              void* d_out, int out_size, void* d_ws, size_t ws_size,
                              hipStream_t stream) {
    const float* corr = (const float*)d_in[0];   // reference_features [B,81,H,W]
    const float* feat = (const float*)d_in[1];   // query_features     [B,C,H,W]
    float* out = (float*)d_out;

    const int H = 96, W = 96, C = 256;
    const int B = in_sizes[0] / (D2 * H * W);    // 16

    dim3 grid(B * H * (W / NT));                 // 9216 workgroups
    corr_transpose_wmma<<<grid, BLK, 0, stream>>>(corr, feat, out, B, C, H, W);
}
